// MultiHeadAttention_14448269984124
// MI455X (gfx1250) — compile-verified, matrix operand order checked by reading, not run on hardware
//
#include <hip/hip_runtime.h>

typedef __attribute__((ext_vector_type(16))) _Float16 v16h;
typedef __attribute__((ext_vector_type(8)))  _Float16 v8h;
typedef __attribute__((ext_vector_type(8)))  float    v8f;

#define NB    32      // batch
#define NSQ   196     // query sequence
#define NFD   2048    // feature dim
#define NL    1000    // labels
#define NE    1024    // embed dim
#define NH    16      // heads
#define NDH   64      // head dim

#if __has_builtin(__builtin_amdgcn_global_load_async_to_lds_b128) && \
    __has_builtin(__builtin_amdgcn_s_wait_asynccnt)
#define USE_ASYNC_LDS 1
typedef int v4i_vec __attribute__((vector_size(16)));
typedef __attribute__((address_space(1))) v4i_vec g_v4i;  // global (AS1)
typedef __attribute__((address_space(3))) v4i_vec l_v4i;  // LDS (AS3)
#else
#define USE_ASYNC_LDS 0
#endif

// 16-bit WMMA A/B operand element mapping (ISA 7.12.2):
// lane L holds matrix row (L & 15); element j of the v16h maps to
// K = kb + j      for j in [0,8)   (VGPR0..3)
// K = kb + j + 8  for j in [8,16)  (VGPR4..7), kb = (L<16) ? 0 : 8
__device__ __forceinline__ int koff(int lane, int j) {
  int kb = (lane & 16) ? 8 : 0;
  return kb + j + (j & 8);
}

// ---------------------------------------------------------------------------
// GEMM: C[M,N] = A[M,K] (f32) * B[N,K]^T (f32) + bias[N]
// out_half!=0 -> write _Float16 to Ch ; else write float to Cf
// block = 256 threads (8 waves). Block tile 128x128, wave tile 32x64,
// K step 32 -> 8 v_wmma_f32_16x16x32_f16 per wave per K step.
// ---------------------------------------------------------------------------
__global__ __launch_bounds__(256) void gemm_bt_f16(
    const float* __restrict__ A, const float* __restrict__ Bw,
    const float* __restrict__ bias, float* __restrict__ Cf,
    _Float16* __restrict__ Ch, int M, int N, int K, int out_half)
{
  __shared__ alignas(16) _Float16 As[128][40];
  __shared__ alignas(16) _Float16 Bs[128][40];

  const int t    = threadIdx.x;
  const int lane = t & 31;
  const int wave = t >> 5;
  const int wm   = wave & 3;   // 0..3 -> 32-row slab
  const int wn   = wave >> 2;  // 0..1 -> 64-col slab
  const int fr   = lane & 15;
  const int hi8  = (lane >> 4) << 3;

  const int m0 = blockIdx.x * 128;
  const int n0 = blockIdx.y * 128;

  // staging: each thread moves 16 contiguous f32 of one row for A and B
  const int srow = t >> 1;         // 0..127
  const int scol = (t & 1) * 16;   // 0 or 16
  const int ar = min(m0 + srow, M - 1);
  const int br = min(n0 + srow, N - 1);

  v8f acc[2][4] = {};

  for (int k0 = 0; k0 < K; k0 += 32) {
    const float* apf = A  + (size_t)ar * K + k0 + scol;
    const float* bpf = Bw + (size_t)br * K + k0 + scol;
    float4 a0 = ((const float4*)apf)[0];
    float4 a1 = ((const float4*)apf)[1];
    float4 a2 = ((const float4*)apf)[2];
    float4 a3 = ((const float4*)apf)[3];
    float4 b0 = ((const float4*)bpf)[0];
    float4 b1 = ((const float4*)bpf)[1];
    float4 b2 = ((const float4*)bpf)[2];
    float4 b3 = ((const float4*)bpf)[3];

    if (k0 + 32 < K) {  // global_prefetch_b8 of the next K tile
      __builtin_prefetch(apf + 32, 0, 1);
      __builtin_prefetch(bpf + 32, 0, 1);
    }

    v8h h0, h1;
    h0[0] = (_Float16)a0.x; h0[1] = (_Float16)a0.y;
    h0[2] = (_Float16)a0.z; h0[3] = (_Float16)a0.w;
    h0[4] = (_Float16)a1.x; h0[5] = (_Float16)a1.y;
    h0[6] = (_Float16)a1.z; h0[7] = (_Float16)a1.w;
    h1[0] = (_Float16)a2.x; h1[1] = (_Float16)a2.y;
    h1[2] = (_Float16)a2.z; h1[3] = (_Float16)a2.w;
    h1[4] = (_Float16)a3.x; h1[5] = (_Float16)a3.y;
    h1[6] = (_Float16)a3.z; h1[7] = (_Float16)a3.w;
    *(v8h*)&As[srow][scol]     = h0;
    *(v8h*)&As[srow][scol + 8] = h1;
    h0[0] = (_Float16)b0.x; h0[1] = (_Float16)b0.y;
    h0[2] = (_Float16)b0.z; h0[3] = (_Float16)b0.w;
    h0[4] = (_Float16)b1.x; h0[5] = (_Float16)b1.y;
    h0[6] = (_Float16)b1.z; h0[7] = (_Float16)b1.w;
    h1[0] = (_Float16)b2.x; h1[1] = (_Float16)b2.y;
    h1[2] = (_Float16)b2.z; h1[3] = (_Float16)b2.w;
    h1[4] = (_Float16)b3.x; h1[5] = (_Float16)b3.y;
    h1[6] = (_Float16)b3.z; h1[7] = (_Float16)b3.w;
    *(v8h*)&Bs[srow][scol]     = h0;
    *(v8h*)&Bs[srow][scol + 8] = h1;
    __syncthreads();

    v16h af[2], bf[4];
#pragma unroll
    for (int s = 0; s < 2; ++s)
#pragma unroll
      for (int j = 0; j < 16; ++j)
        af[s][j] = As[wm * 32 + s * 16 + fr][koff(lane, j)];
#pragma unroll
    for (int s = 0; s < 4; ++s)
#pragma unroll
      for (int j = 0; j < 16; ++j)
        bf[s][j] = Bs[wn * 64 + s * 16 + fr][koff(lane, j)];

#pragma unroll
    for (int ia = 0; ia < 2; ++ia)
#pragma unroll
      for (int ib = 0; ib < 4; ++ib)
        acc[ia][ib] = __builtin_amdgcn_wmma_f32_16x16x32_f16(
            false, af[ia], false, bf[ib], (short)0, acc[ia][ib], false, false);
    __syncthreads();
  }

#pragma unroll
  for (int ib = 0; ib < 4; ++ib) {
    const int nc = n0 + wn * 64 + ib * 16 + fr;
    const float bvv = bias[nc];
#pragma unroll
    for (int ia = 0; ia < 2; ++ia) {
#pragma unroll
      for (int i = 0; i < 8; ++i) {
        const int m = m0 + wm * 32 + ia * 16 + hi8 + i;
        if (m < M) {
          const float x = acc[ia][ib][i] + bvv;
          if (out_half) Ch[(size_t)m * N + nc] = (_Float16)x;
          else          Cf[(size_t)m * N + nc] = x;
        }
      }
    }
  }
}

// ---------------------------------------------------------------------------
// Flash-style attention over the label table.
// grid.x = B*H*4 ; block = 128 (4 waves). Each wave owns 16 query rows of a
// 64-query tile, keeps its own online softmax state + 16x64 f32 context acc.
// K/V chunks staged via GLOBAL_LOAD_ASYNC_TO_LDS_B128 when available.
// ---------------------------------------------------------------------------
__global__ __launch_bounds__(128) void attn_f16(
    const _Float16* __restrict__ Qh, const _Float16* __restrict__ Kh,
    const _Float16* __restrict__ Vh, float* __restrict__ Ctx)
{
  __shared__ alignas(16) _Float16 Ks[32][72];
  __shared__ alignas(16) _Float16 Vs[32][72];
  __shared__ alignas(16) _Float16 Ps[4][16][40];

  const int t    = threadIdx.x;
  const int lane = t & 31;
  const int wave = t >> 5;
  const int fr   = lane & 15;
  const int hi8  = (lane >> 4) << 3;   // C-layout row offset: 0 or 8

  const int qt = blockIdx.x & 3;
  const int h  = (blockIdx.x >> 2) & 15;
  const int b  = blockIdx.x >> 6;

  const int q0 = qt * 64 + wave * 16;  // first query row of this wave

  // Q fragments: 16 x 64 head-dim, as two K=32 WMMA chunks, kept in VGPRs
  v16h qf0, qf1;
  {
    const int qr = min(q0 + fr, NSQ - 1);
    const _Float16* qp = Qh + ((size_t)(b * NSQ + qr)) * NE + h * NDH;
#pragma unroll
    for (int j = 0; j < 16; ++j) {
      const int kk = koff(lane, j);
      qf0[j] = qp[kk];
      qf1[j] = qp[32 + kk];
    }
  }

  float mrow[8], srow[8];
  v8f cacc[4] = {};
#pragma unroll
  for (int i = 0; i < 8; ++i) { mrow[i] = -3.0e38f; srow[i] = 0.0f; }

  const int krr = t >> 2;         // staging row 0..31
  const int kcc = (t & 3) * 16;   // staging col 0,16,32,48 (halves)
  const size_t hoff = (size_t)h * NDH;

  for (int lb = 0; lb < NL; lb += 32) {
    // ---- stage 32 keys/values (f16) into LDS -----------------------------
    const int krow = min(lb + krr, NL - 1);
    const _Float16* kp = Kh + (size_t)krow * NE + hoff + kcc;
    const _Float16* vp = Vh + (size_t)krow * NE + hoff + kcc;
#if USE_ASYNC_LDS
    __builtin_amdgcn_global_load_async_to_lds_b128(
        (g_v4i*)kp,       (l_v4i*)&Ks[krr][kcc],     0, 0);
    __builtin_amdgcn_global_load_async_to_lds_b128(
        (g_v4i*)(kp + 8), (l_v4i*)&Ks[krr][kcc + 8], 0, 0);
    __builtin_amdgcn_global_load_async_to_lds_b128(
        (g_v4i*)vp,       (l_v4i*)&Vs[krr][kcc],     0, 0);
    __builtin_amdgcn_global_load_async_to_lds_b128(
        (g_v4i*)(vp + 8), (l_v4i*)&Vs[krr][kcc + 8], 0, 0);
    __builtin_amdgcn_s_wait_asynccnt(0);
#else
    const float4 k0v = ((const float4*)kp)[0];
    const float4 k1v = ((const float4*)kp)[1];
    const float4 v0v = ((const float4*)vp)[0];
    const float4 v1v = ((const float4*)vp)[1];
    *(float4*)&Ks[krr][kcc]     = k0v;
    *(float4*)&Ks[krr][kcc + 8] = k1v;
    *(float4*)&Vs[krr][kcc]     = v0v;
    *(float4*)&Vs[krr][kcc + 8] = v1v;
#endif
    __syncthreads();

    // ---- scores: 16q x 32keys = 2 C-tiles, each 2 WMMA (K=32 over dh=64) --
    v8f sc0 = {}, sc1 = {};
    {
      v16h kf;
#pragma unroll
      for (int j = 0; j < 16; ++j) kf[j] = Ks[fr][koff(lane, j)];
      sc0 = __builtin_amdgcn_wmma_f32_16x16x32_f16(false, qf0, false, kf,
                                                   (short)0, sc0, false, false);
#pragma unroll
      for (int j = 0; j < 16; ++j) kf[j] = Ks[fr][32 + koff(lane, j)];
      sc0 = __builtin_amdgcn_wmma_f32_16x16x32_f16(false, qf1, false, kf,
                                                   (short)0, sc0, false, false);
#pragma unroll
      for (int j = 0; j < 16; ++j) kf[j] = Ks[16 + fr][koff(lane, j)];
      sc1 = __builtin_amdgcn_wmma_f32_16x16x32_f16(false, qf0, false, kf,
                                                   (short)0, sc1, false, false);
#pragma unroll
      for (int j = 0; j < 16; ++j) kf[j] = Ks[16 + fr][32 + koff(lane, j)];
      sc1 = __builtin_amdgcn_wmma_f32_16x16x32_f16(false, qf1, false, kf,
                                                   (short)0, sc1, false, false);
    }

    // ---- online softmax (per-row state; rows partitioned by wave) --------
    const bool ok0 = (lb + fr) < NL;
    const bool ok1 = (lb + 16 + fr) < NL;
#pragma unroll
    for (int i = 0; i < 8; ++i) {
      float s0 = ok0 ? sc0[i] * 0.125f : -3.0e38f;  // 1/sqrt(64)
      float s1 = ok1 ? sc1[i] * 0.125f : -3.0e38f;
      float mx = fmaxf(s0, s1);
      mx = fmaxf(mx, __shfl_xor(mx, 1, 16));
      mx = fmaxf(mx, __shfl_xor(mx, 2, 16));
      mx = fmaxf(mx, __shfl_xor(mx, 4, 16));
      mx = fmaxf(mx, __shfl_xor(mx, 8, 16));
      const float mn   = fmaxf(mrow[i], mx);
      const float corr = __expf(mrow[i] - mn);
      const float p0   = __expf(s0 - mn);
      const float p1   = __expf(s1 - mn);
      float ps = p0 + p1;
      ps += __shfl_xor(ps, 1, 16);
      ps += __shfl_xor(ps, 2, 16);
      ps += __shfl_xor(ps, 4, 16);
      ps += __shfl_xor(ps, 8, 16);
      srow[i] = srow[i] * corr + ps;
      mrow[i] = mn;
#pragma unroll
      for (int d = 0; d < 4; ++d) cacc[d][i] *= corr;
      // park P (C layout -> LDS) for re-read in A layout
      Ps[wave][hi8 + i][fr]      = (_Float16)p0;
      Ps[wave][hi8 + i][16 + fr] = (_Float16)p1;
    }
    __syncthreads();

    // ---- ctx += P(16x32) * V(32x64): 4 WMMA over dh tiles ----------------
    v16h pf, vf;
#pragma unroll
    for (int j = 0; j < 16; ++j) pf[j] = Ps[wave][fr][koff(lane, j)];
#pragma unroll
    for (int d = 0; d < 4; ++d) {
#pragma unroll
      for (int j = 0; j < 16; ++j) vf[j] = Vs[koff(lane, j)][d * 16 + fr];
      cacc[d] = __builtin_amdgcn_wmma_f32_16x16x32_f16(false, pf, false, vf,
                                                       (short)0, cacc[d],
                                                       false, false);
    }
    __syncthreads();
  }

  // ---- normalize + store context (f32) -----------------------------------
#pragma unroll
  for (int i = 0; i < 8; ++i) {
    const int q = q0 + hi8 + i;
    if (q < NSQ) {
      const float inv = 1.0f / srow[i];
      float* cp = Ctx + ((size_t)(b * NSQ + q)) * NE + h * NDH;
#pragma unroll
      for (int d = 0; d < 4; ++d) cp[d * 16 + fr] = cacc[d][i] * inv;
    }
  }
}

// ---------------------------------------------------------------------------
extern "C" void kernel_launch(void* const* d_in, const int* in_sizes, int n_in,
                              void* d_out, int out_size, void* d_ws, size_t ws_size,
                              hipStream_t stream) {
  const float* features = (const float*)d_in[0];
  const float* labels   = (const float*)d_in[1];
  const float* Wq = (const float*)d_in[2];
  const float* bq = (const float*)d_in[3];
  const float* Wk = (const float*)d_in[4];
  const float* bk = (const float*)d_in[5];
  const float* Wv = (const float*)d_in[6];
  const float* bv = (const float*)d_in[7];
  const float* Wo = (const float*)d_in[8];
  const float* bo = (const float*)d_in[9];
  float* out = (float*)d_out;

  const size_t MQ = (size_t)NB * NSQ;  // 6272
  _Float16* Qh = (_Float16*)d_ws;      // [6272,1024] f16
  _Float16* Kh = Qh + MQ * NE;         // [1000,1024] f16
  _Float16* Vh = Kh + (size_t)NL * NE; // [1000,1024] f16
  float*    ctx = (float*)(Vh + (size_t)NL * NE);  // [6272,1024] f32

  const dim3 blk(256);

  // Q = features @ Wq^T + bq  -> f16   (M=6272 -> 49 tiles, N=1024 -> 8)
  gemm_bt_f16<<<dim3(49, 8), blk, 0, stream>>>(
      features, Wq, bq, nullptr, Qh, (int)MQ, NE, NFD, 1);
  // K = labels @ Wk^T + bk    -> f16   (M=1000 -> 8 tiles)
  gemm_bt_f16<<<dim3(8, 8), blk, 0, stream>>>(
      labels, Wk, bk, nullptr, Kh, NL, NE, NE, 1);
  // V = labels @ Wv^T + bv    -> f16
  gemm_bt_f16<<<dim3(8, 8), blk, 0, stream>>>(
      labels, Wv, bv, nullptr, Vh, NL, NE, NE, 1);

  // attention: softmax(Q K^T / 8) V  -> ctx (f32)
  attn_f16<<<dim3(NB * NH * 4), dim3(128), 0, stream>>>(Qh, Kh, Vh, ctx);

  // out = ctx @ Wo^T + bo     -> f32
  gemm_bt_f16<<<dim3(49, 8), blk, 0, stream>>>(
      ctx, Wo, bo, out, nullptr, (int)MQ, NE, NE, 0);
}
